// CVRP_Decoder_51410758533182
// MI455X (gfx1250) — matrix operand order repaired and checked by reading, compiled, not run on hardware
//
#include <hip/hip_runtime.h>

// ---------------- problem constants ----------------
constexpr int Bc   = 16;
constexpr int Pc   = 100;
constexpr int Nc   = 1000;
constexpr int NPAD = 1024;     // node dim padded to multiple of 32 for WMMA
constexpr int Ec   = 128;
constexpr int Hc   = 8;
constexpr int Dc   = 16;
constexpr int HDc  = 128;      // H*D
constexpr int QD   = 16;
constexpr int DYNc = 3;
constexpr int HIDc = 64;
constexpr int LT   = 4;        // 16-row tiles per wave in k_lazy

typedef __attribute__((ext_vector_type(16))) _Float16 v16h;
typedef __attribute__((ext_vector_type(8)))  float    v8f;

#if __has_builtin(__builtin_amdgcn_tensor_load_to_lds)
#define USE_TDM 1
typedef __attribute__((ext_vector_type(4))) unsigned int u32x4;
typedef __attribute__((ext_vector_type(8))) int i32x8;
typedef __attribute__((ext_vector_type(4))) int i32x4;
#else
#define USE_TDM 0
#endif

__device__ __forceinline__ int laneid() { return (int)(threadIdx.x & 31); }

__device__ __forceinline__ v8f wmma_f16(v16h a, v16h b, v8f c) {
  return __builtin_amdgcn_wmma_f32_16x16x32_f16(false, a, false, b, (short)0, c,
                                                false, false);
}

// A-operand (16x32 f16) from LDS row-major f16 tile [16][ld]; 2x ds_load_b128.
__device__ __forceinline__ v16h lds_a16x32(const _Float16* t, int ld, int k0) {
  int lane = laneid();
  int m    = lane & 15;
  int koff = (lane & 16) ? 16 : 0;
  v16h a;
#pragma unroll
  for (int e = 0; e < 16; ++e) {
    int k = k0 + koff + e;
    a[e] = t[m * ld + k];
  }
  return a;
}

// A-operand from an f32 LDS tile (converts on the fly; 4x ds_load_b128 + cvt).
__device__ __forceinline__ v16h lds_a16x32_f32(const float* t, int ld, int k0) {
  int lane = laneid();
  int m    = lane & 15;
  int koff = (lane & 16) ? 16 : 0;
  v16h a;
#pragma unroll
  for (int e = 0; e < 16; ++e) {
    int k = k0 + koff + e;
    a[e] = (_Float16)t[m * ld + k];
  }
  return a;
}

// Swizzled B-operand storage: per (ks,nt) tile, lane l owns 16 contiguous
// halfs = B[k = ks*32 + (l&16) + e][n = nt*16 + (l&15)].
__device__ __forceinline__ int bswz_idx(int k, int n, int ntiles) {
  int nt = n >> 4, nl = n & 15;
  int ks = k >> 5, kr = k & 31;
  int l = (kr & 16) | nl, e = kr & 15;
  return (((ks * ntiles + nt) * 32) + l) * 16 + e;
}
__device__ __forceinline__ v16h lds_bswz(const _Float16* t, int ntiles, int ks, int nt) {
  return *(const v16h*)(t + ((size_t)((ks * ntiles + nt) * 32 + laneid())) * 16);
}

// ---------------- kernel 1: q, FiLM gamma, qf ----------------
__global__ void __launch_bounds__(128) k_prep_q(
    const float* __restrict__ enc_last, const float* __restrict__ loadv,
    const float* __restrict__ Wq, const float* __restrict__ fW1,
    const float* __restrict__ fb1, const float* __restrict__ fW2,
    const float* __restrict__ fb2, const float* __restrict__ lqW,
    const float* __restrict__ lqb,
    _Float16* __restrict__ qT, _Float16* __restrict__ qf16)
{
  __shared__ float qin[132];
  __shared__ float g1[132];
  int bp = blockIdx.x, b = bp / Pc, p = bp % Pc;
  int t = threadIdx.x;
  if (t < Ec) qin[t] = enc_last[(size_t)(b * Pc + p) * Ec + t];
  if (t == 0) qin[Ec] = loadv[b * Pc + p];
  __syncthreads();
  for (int j = t; j < Ec + 1; j += 128) {
    float acc = fb1[j];
    for (int e = 0; e < Ec + 1; ++e) acc = fmaf(qin[e], fW1[e * (Ec + 1) + j], acc);
    g1[j] = fmaxf(acc, 0.f);
  }
  __syncthreads();
  {
    int j = t;
    float qv = 0.f, gm = fb2[j];
    for (int e = 0; e < Ec + 1; ++e) {
      qv = fmaf(qin[e], Wq[e * HDc + j], qv);
      gm = fmaf(g1[e],  fW2[e * HDc + j], gm);
    }
    float gamma = 2.f / (1.f + __expf(-gm));
    float qs = qv * gamma * 0.25f;           // fold 1/sqrt(D), D=16
    int h = j >> 4, d = j & 15;
    qT[(size_t)((b * Hc + h) * Pc + p) * Dc + d] = (_Float16)qs;
  }
  if (t < QD) {
    float acc = lqb[t];
    for (int e = 0; e < Ec + 1; ++e) acc = fmaf(qin[e], lqW[e * QD + t], acc);
    qf16[(size_t)(b * Pc + p) * QD + t] = (_Float16)acc;
  }
}

// ---------------- kernel 2: K/V projections (WMMA, TDM-staged A) ----------
// K   : (B,H,NPAD,D) f16  -> score B-tile: lanes 0-15 load one v16h, 16-31 zero
// Vt  : (B,H,D,NPAD) f16  -> PV B-tile: contiguous v16h per lane
// encB: (B, 64 ntiles, 4 ks, 32 lanes, 16) f16 pre-swizzled enc^T B operand
__global__ void __launch_bounds__(128) k_prep_kv(
    const float* __restrict__ enc, const float* __restrict__ Wk,
    const float* __restrict__ Wv,
    _Float16* __restrict__ K, _Float16* __restrict__ Vt,
    _Float16* __restrict__ encB)
{
  __shared__ _Float16 __attribute__((aligned(32))) Wkh[Ec * HDc];
  __shared__ _Float16 __attribute__((aligned(32))) Wvh[Ec * HDc];
  __shared__ float    __attribute__((aligned(32))) Af[4][16 * Ec];
  int t = threadIdx.x;
  for (int s = t; s < Ec * HDc; s += 128) {
    int k = s >> 7, n = s & 127;
    int dst = bswz_idx(k, n, 8);
    Wkh[dst] = (_Float16)Wk[s];
    Wvh[dst] = (_Float16)Wv[s];
  }
  __syncthreads();
  int b = blockIdx.x >> 3;
  int tile8 = blockIdx.x & 7;
  int wave = t >> 5, lane = t & 31;
  for (int rep = 0; rep < 2; ++rep) {
    int nt = tile8 * 8 + wave * 2 + rep;   // node tile 0..63
    int n0 = nt * 16;
    float* A = Af[wave];
#if USE_TDM
    {
      // Tensor Data Mover: copy a 128x16 f32 tile (row-major) to LDS.
      // tensor_dim1 = remaining valid rows so OOB rows zero-fill padding.
      unsigned long long gaddr =
          (unsigned long long)(size_t)(const void*)(enc + ((size_t)b * Nc + n0) * Ec);
      unsigned int lds_off = (unsigned int)(size_t)(void*)A;
      int rows = Nc - n0; rows = rows < 0 ? 0 : (rows > 16 ? 16 : rows);
      u32x4 g0;
      g0[0] = 1u;                                    // count=1 (user descriptor)
      g0[1] = lds_off;                               // lds_addr [63:32]
      g0[2] = (unsigned int)(gaddr & 0xffffffffu);   // global_addr lo
      g0[3] = (unsigned int)((gaddr >> 32) & 0x01ffffffu) | (2u << 30); // hi | type=2
      i32x8 g1;
      g1[0] = (int)(2u << 16);                       // data_size = 4B
      g1[1] = (int)(((unsigned)Ec & 0xffffu) << 16); // tensor_dim0 lo -> bits 63:48
      g1[2] = (int)((((unsigned)Ec >> 16) & 0xffffu) |
                    (((unsigned)rows & 0xffffu) << 16)); // dim0 hi | dim1 lo
      g1[3] = (int)(((unsigned)Ec) << 16);           // dim1 hi=0 | tile_dim0=128
      g1[4] = 16;                                    // tile_dim1=16, tile_dim2=0
      g1[5] = Ec;                                    // tensor_dim0_stride lo
      g1[6] = 0;
      g1[7] = 0;
      i32x4 z4 = {};
#if __clang_major__ >= 23
      i32x8 z8 = {};
      __builtin_amdgcn_tensor_load_to_lds(g0, g1, z4, z4, z8, 0);
#else
      __builtin_amdgcn_tensor_load_to_lds(g0, g1, z4, z4, 0);
#endif
      __builtin_amdgcn_s_wait_tensorcnt((short)0);
    }
#else
    for (int i = lane; i < 16 * Ec; i += 32) {
      int r = i >> 7, c = i & 127;
      int n = n0 + r;
      A[i] = (n < Nc) ? enc[((size_t)b * Nc + n) * Ec + c] : 0.f;
    }
#endif
    // pre-swizzled enc^T B tiles for the pointer GEMM
#pragma unroll
    for (int ks = 0; ks < 4; ++ks) {
      const float* src = A + (lane & 15) * Ec + ks * 32 + (lane & 16);
      v16h v;
#pragma unroll
      for (int e = 0; e < 16; ++e) v[e] = (_Float16)src[e];
      *(v16h*)(encB + ((((size_t)b * 64 + nt) * 4 + ks) * 32 + lane) * 16) = v;
    }
    v16h a[4];
#pragma unroll
    for (int ks = 0; ks < 4; ++ks) a[ks] = lds_a16x32_f32(A, Ec, ks * 32);
#pragma unroll
    for (int h = 0; h < Hc; ++h) {
      v8f ck = {}; v8f cv = {};
#pragma unroll
      for (int ks = 0; ks < 4; ++ks) {
        ck = wmma_f16(a[ks], lds_bswz(Wkh, 8, ks, h), ck);
        cv = wmma_f16(a[ks], lds_bswz(Wvh, 8, ks, h), cv);
      }
      int d = lane & 15;
      int rhi = (lane & 16) ? 8 : 0;
#pragma unroll
      for (int i = 0; i < 8; ++i) {
        int n = n0 + i + rhi;
        K [((size_t)(b * Hc + h) * NPAD + n) * Dc + d]  = (_Float16)ck[i];
        Vt[((size_t)(b * Hc + h) * Dc + d) * NPAD + n]  = (_Float16)cv[i];
      }
    }
  }
}

// ---------------- kernel 3: -inf in padded mask columns ----------------
__global__ void __launch_bounds__(256) k_pad_mask(float* __restrict__ mask)
{
  int i = blockIdx.x * 256 + threadIdx.x;
  int tot = Bc * Pc * (NPAD - Nc);
  if (i < tot) {
    int bp = i / (NPAD - Nc), c = i % (NPAD - Nc);
    mask[(size_t)bp * NPAD + Nc + c] = -1e30f;
  }
}

// ---------------- kernel 4: lazy-mask MLP (dominant FLOPs, WMMA) ----------
// Feature order permuted to [qf(16), dyn(3), pad(13)] (W1 rows permuted to match)
// so the per-row x-stage is 1 v16h store + 3 scalar halfs.
__global__ void __launch_bounds__(256) k_lazy(
    const float* __restrict__ dyn, const _Float16* __restrict__ qf16,
    const float* __restrict__ W1, const float* __restrict__ b1,
    const float* __restrict__ W2, const float* __restrict__ b2,
    const float* __restrict__ W3, const float* __restrict__ b3,
    const float* __restrict__ ninf, float* __restrict__ mask)
{
  __shared__ _Float16 __attribute__((aligned(32))) W1h[32 * HIDc];     // swizzled, k-permuted
  __shared__ _Float16 __attribute__((aligned(32))) W2h[HIDc * HIDc];   // swizzled
  __shared__ _Float16 __attribute__((aligned(32))) W3sw[2 * 32 * 16];  // w3 in col 0 B tiles
  __shared__ float b1s[HIDc], b2s[HIDc];
  __shared__ _Float16 __attribute__((aligned(32))) xst[8][16 * 32];
  __shared__ _Float16 __attribute__((aligned(32))) hst[8][16 * HIDc];
  int t = threadIdx.x;
  for (int s = t; s < 32 * HIDc; s += 256) {
    int ko = s >> 6, n = s & 63;                    // source row (original k)
    int kn = (ko < DYNc) ? (QD + ko) : (ko - DYNc); // permuted row; ko>=19 -> kn=ko (pad)
    if (ko >= DYNc + QD) kn = ko;
    W1h[bswz_idx(kn, n, 4)] = (ko < DYNc + QD) ? (_Float16)W1[s] : (_Float16)0.f;
  }
  for (int s = t; s < HIDc * HIDc; s += 256) {
    int k = s >> 6, n = s & 63;
    W2h[bswz_idx(k, n, 4)] = (_Float16)W2[s];
  }
  for (int s = t; s < 2 * 32 * 16; s += 256) {
    int ks = s >> 9, l = (s >> 4) & 31, e = s & 15;
    W3sw[s] = ((l & 15) == 0) ? (_Float16)W3[ks * 32 + (l & 16) + e] : (_Float16)0.f;
  }
  if (t < HIDc) { b1s[t] = b1[t]; b2s[t] = b2[t]; }
  __syncthreads();
  float b3v = b3[0];
  int wave = t >> 5, lane = t & 31;
  long total = (long)Bc * Pc * Nc;
  _Float16* xs = xst[wave];
  _Float16* hs = hst[wave];
  // zero the pad columns of the x-stage once per wave
  for (int i = lane; i < 16 * 32; i += 32) {
    int col = i & 31;
    if (col >= QD + DYNc) xs[i] = (_Float16)0.f;
  }
  int cl = lane & 15;
  int chi = (lane & 16) ? 8 : 0;
  long tilebase = ((long)blockIdx.x * 8 + wave) * LT;

  for (int tt = 0; tt < LT; ++tt) {
    long row0 = (tilebase + tt) * 16;
    if (lane < 16) {                       // lanes 0..15 stage one row each
      long r  = row0 + lane;
      long rc = (r < total) ? r : (total - 1);
      int n = (int)(rc % Nc);
      long bp = rc / Nc;
      v16h qv = *(const v16h*)(qf16 + (size_t)bp * QD);
      *(v16h*)(xs + lane * 32) = qv;       // cols 0..15 = qf
      const float* dynr = dyn + ((size_t)bp * Nc + n) * DYNc;
      xs[lane * 32 + 16] = (_Float16)dynr[0];
      xs[lane * 32 + 17] = (_Float16)dynr[1];
      xs[lane * 32 + 18] = (_Float16)dynr[2];
    }
    v16h a1 = lds_a16x32(xs, 32, 0);
    // layer 1
#pragma unroll
    for (int tl = 0; tl < 4; ++tl) {
      v8f c = {};
      c = wmma_f16(a1, lds_bswz(W1h, 4, 0, tl), c);
      int col = tl * 16 + cl;
      float bb = b1s[col];
#pragma unroll
      for (int i = 0; i < 8; ++i)
        hs[(i + chi) * HIDc + col] = (_Float16)fmaxf(c[i] + bb, 0.f);
    }
    // layer 2
    v16h a2a = lds_a16x32(hs, HIDc, 0);
    v16h a2b = lds_a16x32(hs, HIDc, 32);
#pragma unroll
    for (int tl = 0; tl < 4; ++tl) {
      v8f c = {};
      c = wmma_f16(a2a, lds_bswz(W2h, 4, 0, tl), c);
      c = wmma_f16(a2b, lds_bswz(W2h, 4, 1, tl), c);
      int col = tl * 16 + cl;
      float bb = b2s[col];
#pragma unroll
      for (int i = 0; i < 8; ++i)
        hs[(i + chi) * HIDc + col] = (_Float16)fmaxf(c[i] + bb, 0.f);
    }
    // layer 3 (64->1) as 2 WMMAs, w3 in column 0
    v16h a3a = lds_a16x32(hs, HIDc, 0);
    v16h a3b = lds_a16x32(hs, HIDc, 32);
    v16h w3a = *(const v16h*)(W3sw + (size_t)lane * 16);
    v16h w3b = *(const v16h*)(W3sw + (size_t)(32 + lane) * 16);
    v8f c3 = {};
    c3 = wmma_f16(a3a, w3a, c3);
    c3 = wmma_f16(a3b, w3b, c3);
    if ((lane & 15) == 0) {               // lane 0: rows 0..7, lane 16: rows 8..15
      int rbase = (lane & 16) ? 8 : 0;
#pragma unroll
      for (int i = 0; i < 8; ++i) {
        long r3 = row0 + rbase + i;
        if (r3 < total) {
          float acc = c3[i] + b3v;
          float sp = (acc > 20.f) ? acc : log1pf(__expf(acc));
          long bp3 = r3 / Nc; int n3 = (int)(r3 % Nc);
          mask[(size_t)bp3 * NPAD + n3] = ninf[r3] - sp;   // ALPHA = 1.0
        }
      }
    }
  }
}

// ---------------- kernel 5: flash attention (WMMA, l via ones-WMMA) -------
__global__ void __launch_bounds__(256) k_attn(
    const _Float16* __restrict__ qT, const _Float16* __restrict__ K,
    const _Float16* __restrict__ Vt, const float* __restrict__ mask,
    _Float16* __restrict__ oc)
{
  __shared__ _Float16 __attribute__((aligned(32))) pst[8][16 * 32];
  int t = threadIdx.x;
  int wave = t >> 5, lane = t & 31;
  int g = blockIdx.x * 8 + wave;          // 0..895 = B*H*7
  int b  = g / (Hc * 7);
  int h  = (g / 7) % Hc;
  int p0 = (g % 7) * 16;
  int m = lane & 15;
  int hi = (lane & 16) ? 1 : 0;
  int cl = lane & 15;
  int chi = hi * 8;

  int pr = p0 + m;
  int prc = (pr < Pc) ? pr : (Pc - 1);
  const _Float16* qrow = qT + ((size_t)(b * Hc + h) * Pc + prc) * Dc;
  int koff = hi ? 16 : 0;
  v16h aq;
#pragma unroll
  for (int e = 0; e < 16; ++e) {
    int k = koff + e;
    aq[e] = (k < Dc && pr < Pc) ? qrow[k] : (_Float16)0.f;
  }
  // constant B tile: ones in column 0 -> row-sum accumulator via WMMA
  v16h bones;
#pragma unroll
  for (int e = 0; e < 16; ++e) bones[e] = (cl == 0) ? (_Float16)1.0f : (_Float16)0.f;

  const _Float16* Kb  = K  + (size_t)(b * Hc + h) * NPAD * Dc;
  const _Float16* Vtb = Vt + (size_t)(b * Hc + h) * Dc * NPAD;
  v8f acc = {}, accl = {};
  float mrow[8];
#pragma unroll
  for (int i = 0; i < 8; ++i) mrow[i] = -1e30f;
  _Float16* ps = pst[wave];

  for (int nch = 0; nch < NPAD / 32; ++nch) {
    int n0 = nch * 32;
    if (nch + 2 < NPAD / 32) {
      __builtin_prefetch(Kb + (size_t)(n0 + 64 + cl) * Dc, 0, 0);
      __builtin_prefetch(Vtb + (size_t)cl * NPAD + n0 + 64, 0, 0);
    }
    v16h bk0 = {}, bk1 = {};
    if (!hi) {
      bk0 = *(const v16h*)(Kb + (size_t)(n0 + cl) * Dc);
      bk1 = *(const v16h*)(Kb + (size_t)(n0 + cl + 16) * Dc);
    }
    v8f z0 = {}, z1 = {};
    v8f s0 = wmma_f16(aq, bk0, z0);
    v8f s1 = wmma_f16(aq, bk1, z1);
    float rmax[8];
#pragma unroll
    for (int i = 0; i < 8; ++i) {
      int prow = p0 + i + chi;
      int prowc = (prow < Pc) ? prow : (Pc - 1);
      const float* mr = mask + ((size_t)b * Pc + prowc) * NPAD + n0;
      s0[i] += mr[cl];
      s1[i] += mr[cl + 16];
      rmax[i] = fmaxf(s0[i], s1[i]);
    }
#pragma unroll
    for (int dd = 1; dd < 16; dd <<= 1) {
#pragma unroll
      for (int i = 0; i < 8; ++i)
        rmax[i] = fmaxf(rmax[i], __shfl_xor(rmax[i], dd, 32));
    }
#pragma unroll
    for (int i = 0; i < 8; ++i) {
      float mn  = fmaxf(mrow[i], rmax[i]);
      float sc  = __expf(mrow[i] - mn);
      float pv0 = __expf(s0[i] - mn);
      float pv1 = __expf(s1[i] - mn);
      mrow[i] = mn;
      acc[i]  *= sc;
      accl[i] *= sc;
      ps[(i + chi) * 32 + cl]      = (_Float16)pv0;
      ps[(i + chi) * 32 + cl + 16] = (_Float16)pv1;
    }
    v16h ap = lds_a16x32(ps, 32, 0);
    v16h bv = *(const v16h*)(Vtb + (size_t)cl * NPAD + n0 + (hi ? 16 : 0));
    acc  = wmma_f16(ap, bv, acc);
    accl = wmma_f16(ap, bones, accl);
  }
#pragma unroll
  for (int i = 0; i < 8; ++i) {
    int prow = p0 + i + chi;
    if (prow < Pc) {
      float li = __shfl(accl[i], hi ? 16 : 0, 32);   // broadcast col-0 row sum
      float v = (li > 0.f) ? acc[i] / li : 0.f;
      oc[((size_t)b * Pc + prow) * HDc + h * Dc + cl] = (_Float16)v;
    }
  }
}

// ---------------- kernel 6: W_comb + pointer scores (WMMA) ----------------
__global__ void __launch_bounds__(256) k_comb_ptr(
    const _Float16* __restrict__ oc, const float* __restrict__ Wc,
    const float* __restrict__ bc, const _Float16* __restrict__ encB,
    const float* __restrict__ ninf, float* __restrict__ out)
{
  __shared__ _Float16 __attribute__((aligned(32))) Wch[Ec * Ec];   // swizzled
  __shared__ _Float16 __attribute__((aligned(32))) mh16[16 * Ec];
  int t = threadIdx.x;
  for (int s = t; s < Ec * Ec; s += 256) {
    int k = s >> 7, n = s & 127;
    Wch[bswz_idx(k, n, 8)] = (_Float16)Wc[s];
  }
  int b  = blockIdx.x / 7;
  int p0 = (blockIdx.x % 7) * 16;
  int wave = t >> 5, lane = t & 31;
  int m = lane & 15, hi = (lane & 16) ? 1 : 0, cl = lane & 15, chi = hi * 8;

  int pr = p0 + m;
  int prc = (pr < Pc) ? pr : (Pc - 1);
  const _Float16* ocr = oc + ((size_t)b * Pc + prc) * HDc;
  int koff = hi ? 16 : 0;
  v16h aoc[4];
#pragma unroll
  for (int ks = 0; ks < 4; ++ks) {
#pragma unroll
    for (int e = 0; e < 16; ++e) {
      int k = ks * 32 + koff + e;
      aoc[ks][e] = (pr < Pc) ? ocr[k] : (_Float16)0.f;
    }
  }
  __syncthreads();
  {
    int ct = wave;   // 8 waves, 8 column tiles of mh
    v8f c = {};
#pragma unroll
    for (int ks = 0; ks < 4; ++ks)
      c = wmma_f16(aoc[ks], lds_bswz(Wch, 8, ks, ct), c);
    int col = ct * 16 + cl;
    float bb = bc[col];
#pragma unroll
    for (int i = 0; i < 8; ++i)
      mh16[(i + chi) * Ec + col] = (_Float16)(c[i] + bb);
  }
  __syncthreads();
  v16h amh[4];
#pragma unroll
  for (int ks = 0; ks < 4; ++ks) amh[ks] = lds_a16x32(mh16, Ec, ks * 32);
  const _Float16* encBb = encB + (size_t)b * 64 * 4 * 512;
  for (int nt = wave; nt < 64; nt += 8) {
    int node = nt * 16 + cl;
    v8f c = {};
#pragma unroll
    for (int ks = 0; ks < 4; ++ks) {
      v16h bt = *(const v16h*)(encBb + (((size_t)nt * 4 + ks) * 32 + lane) * 16);
      c = wmma_f16(amh[ks], bt, c);
    }
    if (node < Nc) {
#pragma unroll
      for (int i = 0; i < 8; ++i) {
        int prow = p0 + i + chi;
        if (prow < Pc) {
          float s = c[i] * 0.08838834764831845f;     // 1/sqrt(128)
          float v = 10.f * tanhf(s) + ninf[((size_t)b * Pc + prow) * Nc + node];
          out[((size_t)b * Pc + prow) * Nc + node] = v;
        }
      }
    }
  }
}

// ---------------- kernel 7: row softmax over N ----------------
__global__ void __launch_bounds__(256) k_softmax(float* __restrict__ out)
{
  __shared__ float red[256];
  size_t base = (size_t)blockIdx.x * Nc;
  int t = threadIdx.x;
  float mx = -1e30f;
  for (int i = t; i < Nc; i += 256) mx = fmaxf(mx, out[base + i]);
  red[t] = mx; __syncthreads();
  for (int s = 128; s > 0; s >>= 1) {
    if (t < s) red[t] = fmaxf(red[t], red[t + s]);
    __syncthreads();
  }
  mx = red[0];
  __syncthreads();
  float sum = 0.f;
  for (int i = t; i < Nc; i += 256) {
    float e = __expf(out[base + i] - mx);
    out[base + i] = e;
    sum += e;
  }
  red[t] = sum; __syncthreads();
  for (int s = 128; s > 0; s >>= 1) {
    if (t < s) red[t] += red[t + s];
    __syncthreads();
  }
  float inv = 1.f / red[0];
  for (int i = t; i < Nc; i += 256) out[base + i] *= inv;
}

// ---------------- host launcher ----------------
extern "C" void kernel_launch(void* const* d_in, const int* in_sizes, int n_in,
                              void* d_out, int out_size, void* d_ws, size_t ws_size,
                              hipStream_t stream) {
  const float* enc      = (const float*)d_in[0];
  const float* enc_last = (const float*)d_in[1];
  const float* loadv    = (const float*)d_in[2];
  const float* ninf     = (const float*)d_in[3];
  const float* dyn      = (const float*)d_in[4];
  const float* Wq       = (const float*)d_in[5];
  const float* Wk       = (const float*)d_in[6];
  const float* Wv       = (const float*)d_in[7];
  const float* Wc       = (const float*)d_in[8];
  const float* bcv      = (const float*)d_in[9];
  const float* fW1      = (const float*)d_in[10];
  const float* fb1      = (const float*)d_in[11];
  const float* fW2      = (const float*)d_in[12];
  const float* fb2      = (const float*)d_in[13];
  const float* lqW      = (const float*)d_in[14];
  const float* lqb      = (const float*)d_in[15];
  const float* lW1      = (const float*)d_in[16];
  const float* lb1      = (const float*)d_in[17];
  const float* lW2      = (const float*)d_in[18];
  const float* lb2      = (const float*)d_in[19];
  const float* lW3      = (const float*)d_in[20];
  const float* lb3      = (const float*)d_in[21];
  float* out = (float*)d_out;

  // workspace carve-up (~20 MB)
  _Float16* qT  = (_Float16*)d_ws;                        // B*H*P*D
  _Float16* qf  = qT + (size_t)Bc * Hc * Pc * Dc;         // B*P*QD
  _Float16* K   = qf + (size_t)Bc * Pc * QD;              // B*H*NPAD*D
  _Float16* Vt  = K  + (size_t)Bc * Hc * NPAD * Dc;       // B*H*D*NPAD
  float* mask   = (float*)(Vt + (size_t)Bc * Hc * Dc * NPAD);  // B*P*NPAD
  _Float16* oc  = (_Float16*)(mask + (size_t)Bc * Pc * NPAD);  // B*P*HD
  _Float16* encB = oc + (size_t)Bc * Pc * HDc;            // B*64*4*512

  k_prep_q<<<Bc * Pc, 128, 0, stream>>>(enc_last, loadv, Wq, fW1, fb1, fW2, fb2,
                                        lqW, lqb, qT, qf);
  k_prep_kv<<<Bc * 8, 128, 0, stream>>>(enc, Wk, Wv, K, Vt, encB);
  k_pad_mask<<<(Bc * Pc * (NPAD - Nc) + 255) / 256, 256, 0, stream>>>(mask);
  k_lazy<<<(Bc * Pc * Nc) / (128 * LT), 256, 0, stream>>>(dyn, qf, lW1, lb1, lW2,
                                                          lb2, lW3, lb3, ninf, mask);
  k_attn<<<(Bc * Hc * 7) / 8, 256, 0, stream>>>(qT, K, Vt, mask, oc);
  k_comb_ptr<<<Bc * 7, 256, 0, stream>>>(oc, Wc, bcv, encB, ninf, out);
  k_softmax<<<Bc * Pc, 256, 0, stream>>>(out);
}
